// GRU_77034533421530
// MI455X (gfx1250) — matrix operand order repaired and checked by reading, not compiled, not run on hardware
//
#include <hip/hip_runtime.h>
#include <hip/hip_bf16.h>
#include <stdint.h>

typedef _Float16 v16h __attribute__((ext_vector_type(16)));
typedef float    v8f  __attribute__((ext_vector_type(8)));
typedef unsigned int u32x4 __attribute__((ext_vector_type(4)));
typedef int          i32x8 __attribute__((ext_vector_type(8)));
typedef int          i32x4 __attribute__((ext_vector_type(4)));

#define GRU_B 256
#define GRU_T 512
#define GRU_I 64
#define GRU_H 128
#define G3H   384   // 3*H (gate order r,z,n)

// ---------------------------------------------------------------------------
// WMMA fragment loaders (CDNA5 16-bit layouts, ISA 7.12.2)
// A 16x32:  lane<16 holds row=lane,    K = k0+{0..7, 16..23}
//           lane>=16 holds row=lane-16, K = k0+{8..15, 24..31}
// B 32x16:  lane<16 holds col=lane,    K = k0+{0..7, 16..23}
//           lane>=16 holds col=lane-16, K = k0+{8..15, 24..31}
// ---------------------------------------------------------------------------
template<typename T>
__device__ __forceinline__ v16h load_A_frag(const T* __restrict__ A, int lda,
                                            long long rowBase, int kBase, int lane) {
  long long r = rowBase + (lane & 15);
  int k0 = kBase + ((lane & 16) ? 8 : 0);
  const T* p = A + r * (long long)lda + k0;
  v16h f;
#pragma unroll
  for (int i = 0; i < 8; ++i) f[i]     = (_Float16)p[i];
#pragma unroll
  for (int i = 0; i < 8; ++i) f[8 + i] = (_Float16)p[16 + i];
  return f;
}

__device__ __forceinline__ v16h load_B_frag(const float* __restrict__ W, int ldw,
                                            int colBase, int kBase, int lane) {
  // B = W^T : column n of B is row n of row-major W -> contiguous K per lane
  int n  = colBase + (lane & 15);
  int k0 = kBase + ((lane & 16) ? 8 : 0);
  const float* p = W + (long long)n * ldw + k0;
  v16h f;
#pragma unroll
  for (int i = 0; i < 8; ++i) f[i]     = (_Float16)p[i];
#pragma unroll
  for (int i = 0; i < 8; ++i) f[8 + i] = (_Float16)p[16 + i];
  return f;
}

// fast transcendentals: single v_exp_f32 + v_rcp_f32, no IEEE divide sequence
__device__ __forceinline__ float fast_sigmoid(float x) {
  return __builtin_amdgcn_rcpf(1.0f + __expf(-x));
}
__device__ __forceinline__ float fast_tanh(float x) {
  float e = __expf(-2.0f * x);                      // tanh = (1-e)/(1+e)
  return (1.0f - e) * __builtin_amdgcn_rcpf(1.0f + e);
}

// ---------------------------------------------------------------------------
// TDM: issue a 2D tensor_load_to_lds of a 16 x 384(f16) tile of xw.
// Rows are batch entries (stride T*384 elements), row length 384 f16.
// clang-23 toolchain: 6-arg builtin (g0 u32x4, g1 i32x8, g2 i32x4, g3 i32x4,
// extra i32x8, cpol i32). Descriptor packed per ISA 8.3/8.4.
// ---------------------------------------------------------------------------
__device__ __forceinline__ void tdm_load_xw_tile(const _Float16* gbase, unsigned ldsOff) {
  unsigned long long ga = (unsigned long long)(uintptr_t)gbase;
  u32x4 g0;
  g0[0] = 1u;                                         // count=1, user descriptor
  g0[1] = ldsOff;                                     // lds_addr (bytes)
  g0[2] = (unsigned)(ga & 0xffffffffu);               // global_addr[31:0]
  g0[3] = (unsigned)((ga >> 32) & 0x01ffffffu) | (2u << 30);  // addr[56:32], type=2
  const unsigned td0 = G3H;                           // tensor_dim0
  const unsigned td1 = 16;                            // tensor_dim1
  const unsigned long long s0 = (unsigned long long)GRU_T * G3H;  // dim0 stride
  i32x8 g1;
  g1[0] = (int)(1u << 16);                            // wg_mask=0, data_size=2B
  g1[1] = (int)((td0 & 0xffffu) << 16);               // barrier_addr=0, td0 lo
  g1[2] = (int)(((td0 >> 16) & 0xffffu) | ((td1 & 0xffffu) << 16));
  g1[3] = (int)(((td1 >> 16) & 0xffffu) | ((unsigned)G3H << 16)); // tile_dim0=384
  g1[4] = (int)16;                                    // tile_dim1=16, tile_dim2=0
  g1[5] = (int)(unsigned)(s0 & 0xffffffffu);
  g1[6] = (int)(unsigned)((s0 >> 32) & 0xffffu);      // dim1_stride=0
  g1[7] = 0;
  i32x4 z4 = {0, 0, 0, 0};
  i32x8 z8 = {0, 0, 0, 0, 0, 0, 0, 0};
  __builtin_amdgcn_tensor_load_to_lds(g0, g1, z4, z4, z8, 0);
}

// ---------------------------------------------------------------------------
// Input projection: out[r, 0:384] = A[r, 0:in_dim] * W^T + bias   (f16 out)
// ---------------------------------------------------------------------------
template<int A_IS_F16>
__global__ __launch_bounds__(256)
void proj_kernel(const void* __restrict__ Av, const float* __restrict__ W,
                 const float* __restrict__ bias, _Float16* __restrict__ out,
                 int rows, int in_dim) {
  const int lane = threadIdx.x & 31;
  const int wave = threadIdx.x >> 5;
  long long tile = (long long)blockIdx.x * (blockDim.x >> 5) + wave;
  const int nColTiles = G3H / 16;
  int colTile = (int)(tile % nColTiles);
  long long rowBase = (tile / nColTiles) * 16;
  if (rowBase >= rows) return;                 // wave-uniform
  int colBase = colTile * 16;

  v8f acc = {};
  for (int kc = 0; kc < in_dim; kc += 32) {
    v16h a;
    if constexpr (A_IS_F16) a = load_A_frag((const _Float16*)Av, in_dim, rowBase, kc, lane);
    else                    a = load_A_frag((const float*)Av,    in_dim, rowBase, kc, lane);
    v16h b = load_B_frag(W, in_dim, colBase, kc, lane);
    acc = __builtin_amdgcn_wmma_f32_16x16x32_f16(false, a, false, b, (short)0, acc,
                                                 false, false);
  }
  int n = colBase + (lane & 15);
  int mBase = (lane & 16) ? 8 : 0;
  float bv = bias[n];
#pragma unroll
  for (int v = 0; v < 8; ++v)
    out[(rowBase + mBase + v) * (long long)G3H + n] = (_Float16)(acc[v] + bv);
}

// ---------------------------------------------------------------------------
// Recurrence. One block = 16 batch rows x one direction, all 512 steps local.
//  - Whh^T persistent in VGPRs (12 x v16h per wave).
//  - Wave w owns gate-column tiles {w, 8+w, 16+w}  ->  its three WMMA
//    accumulators are exactly r/z/n for hidden cols [16w,16w+16): no LDS
//    exchange of gh needed.
//  - xw staged one step ahead by the Tensor Data Mover into double-buffered
//    LDS; pipelined with s_wait_tensorcnt 1.
// Manual LDS partition (byte offsets are compile-time constants for the D#):
//   [0      .. 24576)  xwS   : 2 x 16 x 384 f16
//   [24576  .. 32768)  hF32  : 16 x 128 f32
//   [32768  .. 36864)  hF16  : 16 x 128 f16
//   [36864  .. 38400)  bhhS  : 384 f32
// ---------------------------------------------------------------------------
#define XWS_OFF   0u
#define XWS_BYTES 12288u
#define HF32_OFF  24576u
#define HF16_OFF  32768u
#define BHH_OFF   36864u
#define SMEM_SIZE 38400u

__global__ __launch_bounds__(256)
void gru_rec_kernel(const _Float16* __restrict__ xw0, const _Float16* __restrict__ xw1,
                    const float* __restrict__ Whh0, const float* __restrict__ Whh1,
                    const float* __restrict__ bhh0, const float* __restrict__ bhh1,
                    _Float16* __restrict__ h_cat,   // [B,T,2H] f16 (or null)
                    float* __restrict__ h_last) {   // [B,2H] f32 (or null)
  const int lane = threadIdx.x & 31;
  const int wave = threadIdx.x >> 5;
  const int dir  = blockIdx.y;                  // 0 = fwd, 1 = bwd
  const int b0   = blockIdx.x * 16;
  const _Float16* xw  = dir ? xw1  : xw0;
  const float*    Whh = dir ? Whh1 : Whh0;
  const float*    bhh = dir ? bhh1 : bhh0;
  const int dirOff = dir * GRU_H;

  __shared__ __align__(16) unsigned char smem[SMEM_SIZE];
  _Float16* xwS  = (_Float16*)(smem + XWS_OFF);   // [2][16][384]
  float*    hF32 = (float*)   (smem + HF32_OFF);  // [16][128]
  _Float16* hF16 = (_Float16*)(smem + HF16_OFF);  // [16][128]
  float*    bhhS = (float*)   (smem + BHH_OFF);   // [384]

  for (int i = threadIdx.x; i < 16 * GRU_H; i += blockDim.x) {
    hF16[i] = (_Float16)0.0f;
    hF32[i] = 0.0f;
  }
  for (int i = threadIdx.x; i < G3H; i += blockDim.x) bhhS[i] = bhh[i];

  // Persistent B fragments: tiles {w, 8+w, 16+w} x 4 K-chunks, loaded once.
  v16h Bf[3][4];
#pragma unroll
  for (int g = 0; g < 3; ++g) {
    int c = (g * 8 + wave) * 16;               // r/z/n slice for cols [16w,16w+16)
#pragma unroll
    for (int kc = 0; kc < 4; ++kc)
      Bf[g][kc] = load_B_frag(Whh, GRU_H, c, kc * 32, lane);
  }

  // TDM: stage xw tile for t = 0
  if (wave == 0) {
    int tt0 = dir ? (GRU_T - 1) : 0;
    tdm_load_xw_tile(xw + ((long long)b0 * GRU_T + tt0) * G3H, XWS_OFF);
  }
  __syncthreads();

  for (int t = 0; t < GRU_T; ++t) {
    const int tt = dir ? (GRU_T - 1 - t) : t;

    // TDM: stage xw tile for t+1 into the other buffer (overlaps the WMMAs)
    if (wave == 0 && t + 1 < GRU_T) {
      int nt = dir ? (GRU_T - 2 - t) : (t + 1);
      tdm_load_xw_tile(xw + ((long long)b0 * GRU_T + nt) * G3H,
                       XWS_OFF + (unsigned)((t + 1) & 1) * XWS_BYTES);
    }

    // A fragments of h from LDS (f16), all 4 K-chunks
    v16h Af[4];
#pragma unroll
    for (int kc = 0; kc < 4; ++kc) {
      int r  = lane & 15;
      int k0 = kc * 32 + ((lane & 16) ? 8 : 0);
#pragma unroll
      for (int i = 0; i < 8; ++i) Af[kc][i]     = hF16[r * GRU_H + k0 + i];
#pragma unroll
      for (int i = 0; i < 8; ++i) Af[kc][8 + i] = hF16[r * GRU_H + k0 + 16 + i];
    }

    // gh slices for this wave's hidden columns: r, z, n accumulators
    v8f accR = {}, accZ = {}, accN = {};
#pragma unroll
    for (int kc = 0; kc < 4; ++kc)
      accR = __builtin_amdgcn_wmma_f32_16x16x32_f16(false, Af[kc], false, Bf[0][kc],
                                                    (short)0, accR, false, false);
#pragma unroll
    for (int kc = 0; kc < 4; ++kc)
      accZ = __builtin_amdgcn_wmma_f32_16x16x32_f16(false, Af[kc], false, Bf[1][kc],
                                                    (short)0, accZ, false, false);
#pragma unroll
    for (int kc = 0; kc < 4; ++kc)
      accN = __builtin_amdgcn_wmma_f32_16x16x32_f16(false, Af[kc], false, Bf[2][kc],
                                                    (short)0, accN, false, false);

    // pipeline: previous step's TDM must have landed (in-order => depth 1)
    if (wave == 0) {
      if (t + 1 < GRU_T) __builtin_amdgcn_s_wait_tensorcnt(1);
      else               __builtin_amdgcn_s_wait_tensorcnt(0);
    }
    __syncthreads();   // A-frag reads done everywhere; xw[t] visible to all

    // Gate math: wave w owns hidden columns [w*16, w*16+16); accumulators are
    // already in C-layout (lane<16 -> M=v, lane>=16 -> M=v+8, N=lane%16).
    {
      int n   = (lane & 15);
      int mB  = (lane & 16) ? 8 : 0;
      int col = wave * 16 + n;
      float br = bhhS[col], bz = bhhS[GRU_H + col], bn = bhhS[2 * GRU_H + col];
      const _Float16* xwT = xwS + (size_t)(t & 1) * 16 * G3H;
#pragma unroll
      for (int v = 0; v < 8; ++v) {
        int m = mB + v;
        float xr = (float)xwT[m * G3H + col];
        float xz = (float)xwT[m * G3H + GRU_H + col];
        float xn = (float)xwT[m * G3H + 2 * GRU_H + col];
        float rg = fast_sigmoid(xr + accR[v] + br);
        float zg = fast_sigmoid(xz + accZ[v] + bz);
        float ng = fast_tanh(xn + rg * (accN[v] + bn));
        float hn = (1.0f - zg) * ng + zg * hF32[m * GRU_H + col];
        hF32[m * GRU_H + col] = hn;
        hF16[m * GRU_H + col] = (_Float16)hn;
        if (h_cat)
          h_cat[((long long)(b0 + m) * GRU_T + tt) * (2 * GRU_H) + dirOff + col] =
              (_Float16)hn;
        if (h_last && tt == GRU_T - 1)
          h_last[(long long)(b0 + m) * (2 * GRU_H) + dirOff + col] = hn;
      }
    }
    __syncthreads();   // h updates visible before next step's A-frag reads
  }
}

// ---------------------------------------------------------------------------
// out[b, :2] = relu(h_last[b, :256]) @ fc_w^T + fc_b
// ---------------------------------------------------------------------------
__global__ void fc_kernel(const float* __restrict__ hl, const float* __restrict__ w,
                          const float* __restrict__ b, float* __restrict__ out) {
  int i = blockIdx.x * blockDim.x + threadIdx.x;
  if (i >= GRU_B) return;
  float a0 = b[0], a1 = b[1];
  for (int k = 0; k < 2 * GRU_H; ++k) {
    float h = fmaxf(hl[i * 2 * GRU_H + k], 0.0f);
    a0 += h * w[k];
    a1 += h * w[2 * GRU_H + k];
  }
  out[i * 2 + 0] = a0;
  out[i * 2 + 1] = a1;
}

extern "C" void kernel_launch(void* const* d_in, const int* in_sizes, int n_in,
                              void* d_out, int out_size, void* d_ws, size_t ws_size,
                              hipStream_t stream) {
  (void)in_sizes; (void)n_in; (void)out_size; (void)ws_size;
  const float* x     = (const float*)d_in[0];
  const float* Wih0f = (const float*)d_in[1];
  const float* Whh0f = (const float*)d_in[2];
  const float* bih0f = (const float*)d_in[3];
  const float* bhh0f = (const float*)d_in[4];
  const float* Wih0b = (const float*)d_in[5];
  const float* Whh0b = (const float*)d_in[6];
  const float* bih0b = (const float*)d_in[7];
  const float* bhh0b = (const float*)d_in[8];
  const float* Wih1f = (const float*)d_in[9];
  const float* Whh1f = (const float*)d_in[10];
  const float* bih1f = (const float*)d_in[11];
  const float* bhh1f = (const float*)d_in[12];
  const float* Wih1b = (const float*)d_in[13];
  const float* Whh1b = (const float*)d_in[14];
  const float* bih1b = (const float*)d_in[15];
  const float* bhh1b = (const float*)d_in[16];
  const float* fc_w  = (const float*)d_in[17];
  const float* fc_b  = (const float*)d_in[18];

  // workspace layout (f16 xw buffers reused across layers): ~269 MB total
  const size_t xwElems = (size_t)GRU_B * GRU_T * G3H;
  const size_t h0Elems = (size_t)GRU_B * GRU_T * 2 * GRU_H;
  _Float16* xwF   = (_Float16*)d_ws;
  _Float16* xwB   = xwF + xwElems;
  _Float16* h0    = xwB + xwElems;
  float*    hlast = (float*)(h0 + h0Elems);

  const int rows = GRU_B * GRU_T;                         // 131072
  const long long nTiles = (long long)(rows / 16) * (G3H / 16);
  const int projBlocks = (int)((nTiles + 7) / 8);         // 8 waves/block, 1 tile/wave

  // layer 0 input projections (A = x fp32)
  proj_kernel<0><<<projBlocks, 256, 0, stream>>>((const void*)x, Wih0f, bih0f, xwF, rows, GRU_I);
  proj_kernel<0><<<projBlocks, 256, 0, stream>>>((const void*)x, Wih0b, bih0b, xwB, rows, GRU_I);

  // layer 0 recurrence, both directions in one grid (16 batch tiles x 2 dirs)
  gru_rec_kernel<<<dim3(GRU_B / 16, 2), 256, 0, stream>>>(xwF, xwB, Whh0f, Whh0b,
                                                          bhh0f, bhh0b, h0, nullptr);

  // layer 1 input projections (A = concatenated hidden, f16, in_dim = 256)
  proj_kernel<1><<<projBlocks, 256, 0, stream>>>((const void*)h0, Wih1f, bih1f, xwF, rows, 2 * GRU_H);
  proj_kernel<1><<<projBlocks, 256, 0, stream>>>((const void*)h0, Wih1b, bih1b, xwB, rows, 2 * GRU_H);

  // layer 1 recurrence: only the last timestep is consumed downstream
  gru_rec_kernel<<<dim3(GRU_B / 16, 2), 256, 0, stream>>>(xwF, xwB, Whh1f, Whh1b,
                                                          bhh1f, bhh1b, nullptr, hlast);

  fc_kernel<<<(GRU_B + 255) / 256, 256, 0, stream>>>(hlast, fc_w, fc_b, (float*)d_out);
}
